// FtMultiheadAttention_83313775608030
// MI455X (gfx1250) — matrix operand order repaired and checked by reading, not compiled, not run on hardware
//
#include <hip/hip_runtime.h>
#include <math.h>

typedef __attribute__((ext_vector_type(16))) __bf16        v16bf;
typedef __attribute__((ext_vector_type(8)))  float         v8f;
typedef __attribute__((ext_vector_type(4)))  unsigned int  u32x4;
typedef __attribute__((ext_vector_type(4)))  int           i32x4;

#if defined(__has_builtin)
#if __has_builtin(__builtin_amdgcn_global_load_async_to_lds_b128) && \
    __has_builtin(__builtin_amdgcn_s_wait_asynccnt)
#define HAVE_ASYNC_LDS 1
#endif
#endif
#ifndef HAVE_ASYNC_LDS
#define HAVE_ASYNC_LDS 0
#endif

#if HAVE_ASYNC_LDS
typedef __attribute__((address_space(1))) i32x4* gvecp;   // global v4i*
typedef __attribute__((address_space(3))) i32x4* lvecp;   // LDS v4i*
#endif

// copy 16 bytes global -> LDS; async (ASYNCcnt-tracked) when available
__device__ __forceinline__ void copy_b128(const unsigned short* g, unsigned short* l) {
#if HAVE_ASYNC_LDS
    __builtin_amdgcn_global_load_async_to_lds_b128((gvecp)(void*)g, (lvecp)(void*)l, 0, 0);
#else
    *reinterpret_cast<u32x4*>(l) = *reinterpret_cast<const u32x4*>(g);
#endif
}
__device__ __forceinline__ void async_fence() {
#if HAVE_ASYNC_LDS
    __builtin_amdgcn_s_wait_asynccnt(0);
#endif
}

__device__ __forceinline__ unsigned short f2bf(float f) {
    unsigned int u = __float_as_uint(f);
    unsigned int r = u + 0x7fffu + ((u >> 16) & 1u);   // round-to-nearest-even
    return (unsigned short)(r >> 16);
}

__device__ __forceinline__ v8f wmma_bf16(v16bf a, v16bf b, v8f c) {
    return __builtin_amdgcn_wmma_f32_16x16x32_bf16(false, a, false, b, (short)0, c,
                                                   false, false);
}

// A-fragment (16x32 bf16): lane M=row, half=lane/16.
// elems 0-7: K = half*8 + 0..7 ; elems 8-15: K = 16 + half*8 + 0..7
__device__ __forceinline__ v16bf ldsA(const unsigned short* p, int row, int stride,
                                      int kbase, int half) {
    union { u32x4 q[2]; v16bf v; } u;
    const unsigned short* b = p + row * stride + kbase + half * 8;
    u.q[0] = *reinterpret_cast<const u32x4*>(b);
    u.q[1] = *reinterpret_cast<const u32x4*>(b + 16);
    return u.v;
}

// B-fragment (32x16 bf16), tile stored [N][K]: lane N=col, half=lane/16.
// elems 0-7: K = half*8 + 0..7 ; elems 8-15: K = 16 + half*8 + 0..7
__device__ __forceinline__ v16bf ldsB(const unsigned short* p, int nrow, int stride,
                                      int kbase, int half) {
    union { u32x4 q[2]; v16bf v; } u;
    const u32x4* b = reinterpret_cast<const u32x4*>(p + nrow * stride + kbase + half * 8);
    u.q[0] = b[0];
    u.q[1] = b[2];
    return u.v;
}

// ---------------- conversion kernels ----------------
__global__ __launch_bounds__(256) void cvt_x(const float* __restrict__ x,
                                             unsigned short* __restrict__ Xb, int n) {
    int i = blockIdx.x * 256 + threadIdx.x;
    if (i < n) Xb[i] = f2bf(x[i]);
}

// WT[n][k], n = type*1024 + h*64 + e ; W_* is [16][1024][64]
__global__ __launch_bounds__(256) void cvt_wqkv(const float* __restrict__ WQ,
                                                const float* __restrict__ WK,
                                                const float* __restrict__ WV,
                                                unsigned short* __restrict__ WT) {
    int i = blockIdx.x * 256 + threadIdx.x;   // 3072*1024
    int n = i >> 10, k = i & 1023;
    int type = n >> 10;
    int he = n & 1023;
    int h = he >> 6, e = he & 63;
    const float* W = (type == 0) ? WQ : (type == 1) ? WK : WV;
    WT[i] = f2bf(W[((size_t)h * 1024 + k) * 64 + e]);
}

// WoT[d][he] ; W_O flat is [he][d]
__global__ __launch_bounds__(256) void cvt_wo(const float* __restrict__ WO,
                                              unsigned short* __restrict__ WoT) {
    int i = blockIdx.x * 256 + threadIdx.x;   // 1024*1024
    int d = i >> 10, he = i & 1023;
    WoT[i] = f2bf(WO[(size_t)he * 1024 + d]);
}

// ---------------- fused QKV projection GEMM ----------------
// Xb [8192][1024] bf16  x  WT [3072][1024] bf16 (N-major), 128x128 tile / WG,
// double-buffered async LDS staging.
// -> Qb/Kb [bh][2048][64] bf16 (Q scaled 0.125), Vt [bh][64][2048] bf16
__global__ __launch_bounds__(256) void qkv_gemm(
    const unsigned short* __restrict__ Xb, const unsigned short* __restrict__ WT,
    const float* __restrict__ bQ, const float* __restrict__ bK, const float* __restrict__ bV,
    unsigned short* __restrict__ Qb, unsigned short* __restrict__ Kb,
    unsigned short* __restrict__ Vt) {
    __shared__ __align__(16) unsigned short xt[2][128 * 40];
    __shared__ __align__(16) unsigned short wt[2][128 * 40];
    const int tid = threadIdx.x, lane = tid & 31, wave = tid >> 5;
    const int half = lane >> 4, ln = lane & 15;
    const int wm = wave & 3, wn = wave >> 2;       // 4x2 wave grid, 32x64 per wave
    const int m0 = blockIdx.x * 128;
    const int n0 = blockIdx.y * 128;

    auto stage = [&](int buf, int k0) {
#pragma unroll
        for (int i = 0; i < 2; ++i) {
            int idx = tid + i * 256;
            int r = idx >> 2, cb = idx & 3;
            copy_b128(Xb + (size_t)(m0 + r) * 1024 + k0 + cb * 8,
                      &xt[buf][r * 40 + cb * 8]);
            copy_b128(WT + (size_t)(n0 + r) * 1024 + k0 + cb * 8,
                      &wt[buf][r * 40 + cb * 8]);
        }
    };

    stage(0, 0);
    async_fence();
    __syncthreads();

    v8f acc[2][4] = {};
    int buf = 0;
    for (int k0 = 0; k0 < 1024; k0 += 32, buf ^= 1) {
        if (k0 + 32 < 1024) stage(buf ^ 1, k0 + 32);   // prefetch next tile
        v16bf a0 = ldsA(xt[buf], wm * 32 + ln, 40, 0, half);
        v16bf a1 = ldsA(xt[buf], wm * 32 + 16 + ln, 40, 0, half);
#pragma unroll
        for (int t = 0; t < 4; ++t) {
            v16bf b = ldsB(wt[buf], wn * 64 + t * 16 + ln, 40, 0, half);
            acc[0][t] = wmma_bf16(a0, b, acc[0][t]);
            acc[1][t] = wmma_bf16(a1, b, acc[1][t]);
        }
        async_fence();
        __syncthreads();
    }

#pragma unroll
    for (int i = 0; i < 2; ++i)
#pragma unroll
        for (int t = 0; t < 4; ++t) {
            int n = n0 + wn * 64 + t * 16 + ln;
            int type = n >> 10;
            int he = n & 1023;
            int h = he >> 6, e = he & 63;
            float bias = ((type == 0) ? bQ : (type == 1) ? bK : bV)[he];
#pragma unroll
            for (int r = 0; r < 8; ++r) {
                int row = m0 + wm * 32 + i * 16 + r + half * 8;
                int b = row >> 11, s = row & 2047;
                float v = acc[i][t][r] + bias;
                if (type == 0)
                    Qb[(((size_t)(b * 16 + h)) * 2048 + s) * 64 + e] = f2bf(v * 0.125f);
                else if (type == 1)
                    Kb[(((size_t)(b * 16 + h)) * 2048 + s) * 64 + e] = f2bf(v);
                else
                    Vt[(((size_t)(b * 16 + h)) * 64 + e) * 2048 + s] = f2bf(v);
            }
        }
}

// ---------------- flash attention ----------------
// per (b,h): Q [2048][64], K [2048][64], V^T [64][2048] -> Ob [b*S][h*64+e] bf16
__global__ __launch_bounds__(256) void attn(
    const unsigned short* __restrict__ Qb, const unsigned short* __restrict__ Kb,
    const unsigned short* __restrict__ Vt, unsigned short* __restrict__ Ob) {
    __shared__ __align__(16) unsigned short kt[2][64 * 72];
    __shared__ __align__(16) unsigned short vt[2][64 * 72];
    __shared__ __align__(16) unsigned short pb[8 * 16 * 72];
    const int tid = threadIdx.x, lane = tid & 31, wave = tid >> 5;
    const int half = lane >> 4, ln = lane & 15;
    const int q0 = blockIdx.x * 128;
    const int bh = blockIdx.y;
    const unsigned short* Qh = Qb + (size_t)bh * 2048 * 64;
    const unsigned short* Kh = Kb + (size_t)bh * 2048 * 64;
    const unsigned short* Vh = Vt + (size_t)bh * 64 * 2048;
    unsigned short* pw = pb + wave * (16 * 72);

    auto stage = [&](int buf, int kv0) {
#pragma unroll
        for (int i = 0; i < 2; ++i) {              // K [kv][e] and V^T [e][kv]
            int idx = tid + i * 256;
            int r = idx >> 3, cb = idx & 7;
            copy_b128(Kh + (size_t)(kv0 + r) * 64 + cb * 8, &kt[buf][r * 72 + cb * 8]);
            copy_b128(Vh + (size_t)r * 2048 + kv0 + cb * 8, &vt[buf][r * 72 + cb * 8]);
        }
    };

    v16bf qf[2];                                   // Q A-fragments, kept in regs
    {
        int M = q0 + wave * 16 + ln;
#pragma unroll
        for (int c = 0; c < 2; ++c) {
            union { u32x4 q[2]; v16bf v; } u;
            const unsigned short* p = Qh + (size_t)M * 64 + c * 32 + half * 8;
            u.q[0] = *reinterpret_cast<const u32x4*>(p);
            u.q[1] = *reinterpret_cast<const u32x4*>(p + 16);
            qf[c] = u.v;
        }
    }
    v8f oacc[4] = {};
    float mrow[8], lrow[8];
#pragma unroll
    for (int r = 0; r < 8; ++r) { mrow[r] = -3.0e38f; lrow[r] = 0.f; }

    stage(0, 0);
    async_fence();
    __syncthreads();

    int buf = 0;
    for (int kv0 = 0; kv0 < 2048; kv0 += 64, buf ^= 1) {
        if (kv0 + 64 < 2048) stage(buf ^ 1, kv0 + 64);   // prefetch next K/V tile

        v8f sa[4];
#pragma unroll
        for (int t = 0; t < 4; ++t) {              // scores: Q x K^T
            v8f z = {};
#pragma unroll
            for (int c = 0; c < 2; ++c) {
                v16bf kf = ldsB(kt[buf], t * 16 + ln, 72, c * 32, half);
                z = wmma_bf16(qf[c], kf, z);
            }
            sa[t] = z;
        }
        // online softmax (row reductions across 16-lane halves)
#pragma unroll
        for (int r = 0; r < 8; ++r) {
            float mt = fmaxf(fmaxf(sa[0][r], sa[1][r]), fmaxf(sa[2][r], sa[3][r]));
#pragma unroll
            for (int off = 1; off < 16; off <<= 1) mt = fmaxf(mt, __shfl_xor(mt, off, 32));
            float mnew = fmaxf(mrow[r], mt);
            float alpha = exp2f((mrow[r] - mnew) * 1.44269504f);
            float psum = 0.f;
#pragma unroll
            for (int t = 0; t < 4; ++t) {
                float p = exp2f((sa[t][r] - mnew) * 1.44269504f);
                sa[t][r] = p;
                psum += p;
            }
#pragma unroll
            for (int off = 1; off < 16; off <<= 1) psum += __shfl_xor(psum, off, 32);
            lrow[r] = lrow[r] * alpha + psum;
            mrow[r] = mnew;
#pragma unroll
            for (int t = 0; t < 4; ++t) oacc[t][r] *= alpha;
        }
        // P: C-layout -> LDS -> A-fragments (wave-private buffer)
#pragma unroll
        for (int t = 0; t < 4; ++t)
#pragma unroll
            for (int r = 0; r < 8; ++r)
                pw[(r + 8 * half) * 72 + t * 16 + ln] = f2bf(sa[t][r]);
        asm volatile("s_wait_dscnt 0" ::: "memory");
#pragma unroll
        for (int c = 0; c < 2; ++c) {              // O += P x V
            v16bf pf = ldsA(pw, ln, 72, c * 32, half);
#pragma unroll
            for (int t = 0; t < 4; ++t) {
                v16bf vf = ldsB(vt[buf], t * 16 + ln, 72, c * 32, half);
                oacc[t] = wmma_bf16(pf, vf, oacc[t]);
            }
        }
        async_fence();
        __syncthreads();
    }
    const int b = bh >> 4, h = bh & 15;
#pragma unroll
    for (int r = 0; r < 8; ++r) {
        float inv = 1.0f / lrow[r];
        int s = q0 + wave * 16 + r + 8 * half;
        size_t rowoff = ((size_t)(b * 2048 + s)) * 1024 + h * 64;
#pragma unroll
        for (int t = 0; t < 4; ++t)
            Ob[rowoff + t * 16 + ln] = f2bf(oacc[t][r] * inv);
    }
}

// ---------------- output projection ----------------
// Ob [8192][1024] bf16  x  WoT [1024][1024] bf16 (N-major) -> out f32 + b_O
__global__ __launch_bounds__(256) void out_gemm(
    const unsigned short* __restrict__ Ob, const unsigned short* __restrict__ WoT,
    const float* __restrict__ bO, float* __restrict__ out) {
    __shared__ __align__(16) unsigned short xt[2][128 * 40];
    __shared__ __align__(16) unsigned short wt[2][128 * 40];
    const int tid = threadIdx.x, lane = tid & 31, wave = tid >> 5;
    const int half = lane >> 4, ln = lane & 15;
    const int wm = wave & 3, wn = wave >> 2;
    const int m0 = blockIdx.x * 128;
    const int n0 = blockIdx.y * 128;

    auto stage = [&](int buf, int k0) {
#pragma unroll
        for (int i = 0; i < 2; ++i) {
            int idx = tid + i * 256;
            int r = idx >> 2, cb = idx & 3;
            copy_b128(Ob + (size_t)(m0 + r) * 1024 + k0 + cb * 8,
                      &xt[buf][r * 40 + cb * 8]);
            copy_b128(WoT + (size_t)(n0 + r) * 1024 + k0 + cb * 8,
                      &wt[buf][r * 40 + cb * 8]);
        }
    };

    stage(0, 0);
    async_fence();
    __syncthreads();

    v8f acc[2][4] = {};
    int buf = 0;
    for (int k0 = 0; k0 < 1024; k0 += 32, buf ^= 1) {
        if (k0 + 32 < 1024) stage(buf ^ 1, k0 + 32);
        v16bf a0 = ldsA(xt[buf], wm * 32 + ln, 40, 0, half);
        v16bf a1 = ldsA(xt[buf], wm * 32 + 16 + ln, 40, 0, half);
#pragma unroll
        for (int t = 0; t < 4; ++t) {
            v16bf b = ldsB(wt[buf], wn * 64 + t * 16 + ln, 40, 0, half);
            acc[0][t] = wmma_bf16(a0, b, acc[0][t]);
            acc[1][t] = wmma_bf16(a1, b, acc[1][t]);
        }
        async_fence();
        __syncthreads();
    }

#pragma unroll
    for (int i = 0; i < 2; ++i)
#pragma unroll
        for (int t = 0; t < 4; ++t) {
            int n = n0 + wn * 64 + t * 16 + ln;
            float bias = bO[n];
#pragma unroll
            for (int r = 0; r < 8; ++r) {
                int row = m0 + wm * 32 + i * 16 + r + half * 8;
                out[(size_t)row * 1024 + n] = acc[i][t][r] + bias;
            }
        }
}

extern "C" void kernel_launch(void* const* d_in, const int* in_sizes, int n_in,
                              void* d_out, int out_size, void* d_ws, size_t ws_size,
                              hipStream_t stream) {
    const float* x  = (const float*)d_in[0];
    const float* WQ = (const float*)d_in[1];
    const float* WK = (const float*)d_in[2];
    const float* WV = (const float*)d_in[3];
    const float* WO = (const float*)d_in[4];
    const float* bQ = (const float*)d_in[5];
    const float* bK = (const float*)d_in[6];
    const float* bV = (const float*)d_in[7];
    const float* bO = (const float*)d_in[8];
    float* out = (float*)d_out;

    char* ws = (char*)d_ws;
    unsigned short* Xb  = (unsigned short*)(ws);                         // 16 MiB
    unsigned short* WT  = (unsigned short*)(ws + (size_t)16 * 1048576);  //  6 MiB
    unsigned short* Qb  = (unsigned short*)(ws + (size_t)22 * 1048576);  // 16 MiB
    unsigned short* Kb  = (unsigned short*)(ws + (size_t)38 * 1048576);  // 16 MiB
    unsigned short* Vt  = (unsigned short*)(ws + (size_t)54 * 1048576);  // 16 MiB
    unsigned short* Ob  = (unsigned short*)(ws + (size_t)70 * 1048576);  // 16 MiB
    unsigned short* WoT = (unsigned short*)(ws + (size_t)86 * 1048576);  //  2 MiB

    cvt_x<<<(8192 * 1024) / 256, 256, 0, stream>>>(x, Xb, 8192 * 1024);
    cvt_wqkv<<<(3072 * 1024) / 256, 256, 0, stream>>>(WQ, WK, WV, WT);
    cvt_wo<<<(1024 * 1024) / 256, 256, 0, stream>>>(WO, WoT);
    qkv_gemm<<<dim3(64, 24), 256, 0, stream>>>(Xb, WT, bQ, bK, bV, Qb, Kb, Vt);
    attn<<<dim3(16, 64), 256, 0, stream>>>(Qb, Kb, Vt, Ob);
    out_gemm<<<dim3(64, 8), 256, 0, stream>>>(Ob, WoT, bO, out);
}
